// EdgeConvNet_87514253623804
// MI455X (gfx1250) — compile-verified, matrix operand order checked by reading, not run on hardware
//
#include <hip/hip_runtime.h>
#include <hip/hip_bf16.h>

typedef _Float16 f16;
typedef __attribute__((ext_vector_type(8)))  _Float16 v8h;
typedef __attribute__((ext_vector_type(16))) _Float16 v16h;
typedef __attribute__((ext_vector_type(8)))  float    v8f;

#define WAVES_PER_BLOCK 8
#define TILE_M 16

// Deterministic float atomic max (order-independent). Init value must be -inf bits.
__device__ __forceinline__ void atomic_max_float(float* addr, float v) {
    if (v >= 0.0f) atomicMax((int*)addr, __float_as_int(v));
    else           atomicMin((unsigned int*)addr, __float_as_uint(v));
}

// Load a 16-half WMMA fragment (A or B^T) from LDS following the CDNA5 16-bit
// striping: lanes 0-15 hold K{base+0..7, base+16..23}, lanes 16-31 hold
// K{base+8..15, base+24..31}.  laneOff = (lane>=16) ? 8 : 0.
__device__ __forceinline__ v16h load_frag(const f16* rowBase, int kBase, int laneOff) {
    const v8h lo = *(const v8h*)(rowBase + kBase + laneOff);
    const v8h hi = *(const v8h*)(rowBase + kBase + 16 + laneOff);
    return __builtin_shufflevector(lo, hi, 0, 1, 2, 3, 4, 5, 6, 7,
                                           8, 9, 10, 11, 12, 13, 14, 15);
}

// Fused EdgeConv layer: per edge e, msg = [x_dst, x_src - x_dst] (f16),
// h = relu(msg @ Wa + ba) @ Wb + bb, atomic-max into agg[dst].
// F_IN == 7  : x_in is f32 [N,7],  K padded 14 -> 32
// F_IN == 64 : x_in is f16 [N,64], K = 128
template <int F_IN>
__global__ __launch_bounds__(256) void edgeconv_wmma_kernel(
    const void* __restrict__ x_in,
    const int* __restrict__ src, const int* __restrict__ dst,
    const float* __restrict__ Wa, const float* __restrict__ ba,
    const float* __restrict__ Wb, const float* __restrict__ bb,
    float* __restrict__ agg, int nEdges)
{
    constexpr int KP    = (F_IN == 7) ? 32 : 128;   // padded K for GEMM1
    constexpr int KREAL = 2 * F_IN;
    constexpr int KS1   = KP / 32;                  // GEMM1 K-steps
    constexpr int AK    = (KP < 64) ? 64 : KP;      // msg buffer also stages 16x64 hidden

    __shared__ __align__(16) f16 sWa[64 * KP];                       // Wa^T: [ch][k]
    __shared__ __align__(16) f16 sWb[64 * 64];                       // Wb^T: [ch][k]
    __shared__ __align__(16) f16 sMsg[WAVES_PER_BLOCK][TILE_M * AK]; // per-wave A tile / hidden
    __shared__ float sBa[64], sBb[64];
    __shared__ int   sDst[WAVES_PER_BLOCK][TILE_M];

    const int tid = threadIdx.x;

    // ---- Weight conversion fp32 -> f16, transposed. Branch-free & coalesced. ----
    if constexpr (KREAL < KP) {               // zero the K padding region
        constexpr int PAD = KP - KREAL;
        for (int i = tid; i < 64 * PAD; i += 256) {
            const int c = i / PAD, k = KREAL + (i % PAD);
            sWa[c * KP + k] = (f16)0.0f;
        }
    }
    for (int i = tid; i < KREAL * 64; i += 256) {   // contiguous reads of Wa[k][c]
        const int k = i >> 6, c = i & 63;
        sWa[c * KP + k] = (f16)Wa[i];
    }
    for (int i = tid; i < 64 * 64; i += 256) {      // contiguous reads of Wb[k][c]
        const int k = i >> 6, c = i & 63;
        sWb[c * 64 + k] = (f16)Wb[i];
    }
    if (tid < 64) { sBa[tid] = ba[tid]; sBb[tid] = bb[tid]; }
    __syncthreads();

    const int wave    = tid >> 5;
    const int lane    = tid & 31;
    const int r       = lane & 15;    // matrix row (edge) / B column within tile
    const int hhalf   = lane >> 4;    // which K-half this lane carries
    const int laneOff = hhalf ? 8 : 0;

    // ---- Hoist ALL loop-invariant B fragments into registers (once per wave). ----
    v16h bfA[KS1][4];   // GEMM1 weights
#pragma unroll
    for (int s = 0; s < KS1; ++s)
#pragma unroll
        for (int t = 0; t < 4; ++t)
            bfA[s][t] = load_frag(sWa + (t * 16 + r) * KP, s * 32, laneOff);
    v16h bfB[2][4];     // GEMM2 weights
#pragma unroll
    for (int s = 0; s < 2; ++s)
#pragma unroll
        for (int t = 0; t < 4; ++t)
            bfB[s][t] = load_frag(sWb + (t * 16 + r) * 64, s * 32, laneOff);

    f16* const msg  = &sMsg[wave][0];   // GEMM1 A tile, stride KP
    f16* const hbuf = &sMsg[wave][0];   // hidden staging, stride 64 (reuse; AK >= 64)
    int* const dvec = &sDst[wave][0];

    const int numTiles     = (nEdges + TILE_M - 1) / TILE_M;
    const int tilesPerIter = gridDim.x * WAVES_PER_BLOCK;

    for (int tile = blockIdx.x * WAVES_PER_BLOCK + wave; tile < numTiles;
         tile += tilesPerIter) {
        const int  e0    = tile * TILE_M;
        const int  e     = e0 + r;
        const bool valid = (e < nEdges);
        const int  si    = valid ? src[e] : 0;
        const int  di    = valid ? dst[e] : 0;
        if (hhalf == 0) dvec[r] = valid ? di : -1;

        // ---- Software pipeline: prefetch next tile's gather rows into cache ----
        {
            const int eN = e + tilesPerIter * TILE_M;
            if (eN < nEdges) {
                const int sn = src[eN];
                const int dn = dst[eN];
                if constexpr (F_IN == 7) {
                    const float* xb = (const float*)x_in;
                    __builtin_prefetch(xb + (long)sn * 7, 0, 0);
                    __builtin_prefetch(xb + (long)dn * 7, 0, 0);
                } else {
                    const f16* xb = (const f16*)x_in;
                    __builtin_prefetch(xb + (long)sn * 64, 0, 0);
                    __builtin_prefetch(xb + (long)dn * 64, 0, 0);
                }
            }
        }

        // ---- Build message tile [16 x KP] in LDS (f16) ----
        if constexpr (F_IN == 7) {
            if (hhalf == 0) {
                const float* xb = (const float*)x_in;
                const float* xd = xb + (long)di * 7;
                const float* xs = xb + (long)si * 7;
                f16* m = msg + r * KP;
#pragma unroll
                for (int f = 0; f < 7; ++f) {
                    const float dv = xd[f];
                    m[f]     = (f16)dv;
                    m[7 + f] = (f16)(xs[f] - dv);
                }
#pragma unroll
                for (int f = 14; f < 32; ++f) m[f] = (f16)0.0f;
            }
        } else {
            const f16* xb = (const f16*)x_in;
            const f16* xd = xb + (long)di * 64 + hhalf * 32;
            const f16* xs = xb + (long)si * 64 + hhalf * 32;
            f16* m1 = msg + r * KP + hhalf * 32;        // x_dst half
            f16* m2 = msg + r * KP + 64 + hhalf * 32;   // (x_src - x_dst) half
#pragma unroll
            for (int f = 0; f < 32; f += 8) {
                const v8h d8 = *(const v8h*)(xd + f);
                const v8h s8 = *(const v8h*)(xs + f);
                *(v8h*)(m1 + f) = d8;
                *(v8h*)(m2 + f) = s8 - d8;
            }
        }
        __builtin_amdgcn_wave_barrier();   // scheduling fence (per-wave LDS tile)

        // ---- GEMM1: [16 x KP] x [KP x 64] -> 4 accumulator tiles of 16x16 f32 ----
        v8f acc[4];
#pragma unroll
        for (int t = 0; t < 4; ++t) acc[t] = {};
#pragma unroll
        for (int s = 0; s < KS1; ++s) {
            const v16h aF = load_frag(msg + r * KP, s * 32, laneOff);
#pragma unroll
            for (int t = 0; t < 4; ++t)
                acc[t] = __builtin_amdgcn_wmma_f32_16x16x32_f16(
                    false, aF, false, bfA[s][t], (short)0, acc[t], false, false);
        }

        // ---- bias + ReLU, stage hidden [16 x 64] as f16 in LDS ----
        {
            const int mBase = hhalf * 8;
#pragma unroll
            for (int t = 0; t < 4; ++t) {
                const int   ch = t * 16 + r;
                const float bv = sBa[ch];
#pragma unroll
                for (int i = 0; i < 8; ++i) {
                    const float v = fmaxf(acc[t][i] + bv, 0.0f);
                    hbuf[(mBase + i) * 64 + ch] = (f16)v;
                }
            }
        }
        __builtin_amdgcn_wave_barrier();

        // ---- GEMM2: [16 x 64] x [64 x 64] ----
#pragma unroll
        for (int t = 0; t < 4; ++t) acc[t] = {};
#pragma unroll
        for (int s = 0; s < 2; ++s) {
            const v16h aF = load_frag(hbuf + r * 64, s * 32, laneOff);
#pragma unroll
            for (int t = 0; t < 4; ++t)
                acc[t] = __builtin_amdgcn_wmma_f32_16x16x32_f16(
                    false, aF, false, bfB[s][t], (short)0, acc[t], false, false);
        }

        // ---- bias + scatter-max into agg[dst][ch] (L2-resident atomics) ----
        {
            const int mBase = hhalf * 8;
#pragma unroll
            for (int t = 0; t < 4; ++t) {
                const int   ch = t * 16 + r;
                const float bv = sBb[ch];
#pragma unroll
                for (int i = 0; i < 8; ++i) {
                    const int d = dvec[mBase + i];
                    if (d >= 0)
                        atomic_max_float(agg + (long)d * 64 + ch, acc[t][i] + bv);
                }
            }
        }
    }
}

__global__ void fill_neg_inf_kernel(unsigned int* __restrict__ p, long n) {
    const long i = (long)blockIdx.x * blockDim.x + threadIdx.x;
    if (i < n) p[i] = 0xFF800000u;   // -inf bits
}

// relu(segment-max) -> f16 node features for next layer's gathers.
// (-inf from empty segments becomes 0, matching PyG fill + relu.)
__global__ void relu_to_f16_kernel(const float* __restrict__ agg,
                                   f16* __restrict__ out, long n) {
    const long i = (long)blockIdx.x * blockDim.x + threadIdx.x;
    if (i < n) out[i] = (f16)fmaxf(agg[i], 0.0f);
}

// Fused relu(segment-max) + final linear [N,64] @ [64,1] + bl.
__global__ void final_linear_kernel(const float* __restrict__ agg,
                                    const float* __restrict__ Wl,
                                    const float* __restrict__ bl,
                                    float* __restrict__ out, int nNodes) {
    const int n = blockIdx.x * blockDim.x + threadIdx.x;
    if (n >= nNodes) return;
    const float* row = agg + (long)n * 64;
    float a = 0.0f;
#pragma unroll
    for (int c = 0; c < 64; ++c) a += fmaxf(row[c], 0.0f) * Wl[c];
    out[n] = a + bl[0];
}

extern "C" void kernel_launch(void* const* d_in, const int* in_sizes, int n_in,
                              void* d_out, int out_size, void* d_ws, size_t ws_size,
                              hipStream_t stream) {
    const float* x    = (const float*)d_in[0];
    const int*   eidx = (const int*)d_in[1];
    const float* W1a  = (const float*)d_in[2];
    const float* b1a  = (const float*)d_in[3];
    const float* W1b  = (const float*)d_in[4];
    const float* b1b  = (const float*)d_in[5];
    const float* W2a  = (const float*)d_in[6];
    const float* b2a  = (const float*)d_in[7];
    const float* W2b  = (const float*)d_in[8];
    const float* b2b  = (const float*)d_in[9];
    const float* Wl   = (const float*)d_in[10];
    const float* bl   = (const float*)d_in[11];
    float* out = (float*)d_out;

    const int nNodes = in_sizes[0] / 7;
    const int nEdges = in_sizes[1] / 2;
    const int* src = eidx;
    const int* dst = eidx + nEdges;

    // Workspace: agg f32 [N,64] (25.6MB, L2-resident), then f16 node feats [N,64].
    char*  ws   = (char*)d_ws;
    float* agg  = (float*)ws;
    f16*   hf16 = (f16*)(ws + (size_t)nNodes * 64 * sizeof(float)); // 16B-aligned

    const long aggN       = (long)nNodes * 64;
    const int  fillBlocks = (int)((aggN + 255) / 256);

    const int numTiles = (nEdges + TILE_M - 1) / TILE_M;
    int edgeBlocks = (numTiles + WAVES_PER_BLOCK - 1) / WAVES_PER_BLOCK;
    if (edgeBlocks > 1536) edgeBlocks = 1536;   // persistent blocks amortize weight->LDS

    // Layer 1
    fill_neg_inf_kernel<<<fillBlocks, 256, 0, stream>>>((unsigned int*)agg, aggN);
    edgeconv_wmma_kernel<7><<<edgeBlocks, 256, 0, stream>>>(
        x, src, dst, W1a, b1a, W1b, b1b, agg, nEdges);
    relu_to_f16_kernel<<<fillBlocks, 256, 0, stream>>>(agg, hf16, aggN);

    // Layer 2 (reuses agg buffer)
    fill_neg_inf_kernel<<<fillBlocks, 256, 0, stream>>>((unsigned int*)agg, aggN);
    edgeconv_wmma_kernel<64><<<edgeBlocks, 256, 0, stream>>>(
        hf16, src, dst, W2a, b2a, W2b, b2b, agg, nEdges);

    // Final linear (fused relu)
    final_linear_kernel<<<(nNodes + 255) / 256, 256, 0, stream>>>(agg, Wl, bl, out, nNodes);
}
